// Informer_53395033424452
// MI455X (gfx1250) — matrix operand order repaired and checked by reading, not compiled, not run on hardware
//
#include <hip/hip_runtime.h>
#include <hip/hip_bf16.h>

typedef __attribute__((ext_vector_type(16))) _Float16 v16h;
typedef __attribute__((ext_vector_type(8)))  _Float16 v8h;
typedef __attribute__((ext_vector_type(4)))  _Float16 v4h;
typedef __attribute__((ext_vector_type(2)))  _Float16 v2h;
typedef __attribute__((ext_vector_type(8)))  float    v8f;

// ---------------------------------------------------------------------------
// WMMA helpers (CDNA5 gfx1250, wave32).
// D = A(16x32 f16) * B(32x16 f16) + C(16x16 f32)
// ---------------------------------------------------------------------------
__device__ inline v8f wmma_f16(v16h a, v16h b, v8f c) {
  return __builtin_amdgcn_wmma_f32_16x16x32_f16(
      /*neg_a=*/false, a, /*neg_b=*/false, b,
      /*c_mod=*/(short)0, c, /*reuse_a=*/false, /*reuse_b=*/false);
}

__device__ inline v16h cat8(v8h lo, v8h hi) {
  return __builtin_shufflevector(lo, hi, 0, 1, 2, 3, 4, 5, 6, 7,
                                 8, 9, 10, 11, 12, 13, 14, 15);
}

// 4 fp32 -> 4 packed fp16 via v_cvt_pk_rtz_f16_f32
__device__ inline v4h cvt4(float a, float b, float c, float d) {
  v2h lo = __builtin_bit_cast(v2h, __builtin_amdgcn_cvt_pkrtz(a, b));
  v2h hi = __builtin_bit_cast(v2h, __builtin_amdgcn_cvt_pkrtz(c, d));
  return __builtin_shufflevector(lo, hi, 0, 1, 2, 3);
}

// A fragment, 16(M) x 32(K), from LDS tile (row-major [m][k], stride lda halves).
// lda, k0 multiples of 8 halves (16B) so both loads are B128-aligned.
__device__ inline v16h load_a_frag(const _Float16* s, int lda, int row0, int k0) {
  const int lane = threadIdx.x & 31;
  const int r    = row0 + (lane & 15);
  const int hoff = (lane >> 4) * 8;
  const _Float16* p = s + (size_t)r * lda + k0 + hoff;
  v8h lo = *(const v8h*)(p);
  v8h hi = *(const v8h*)(p + 16);
  return cat8(lo, hi);
}

// B fragment, 32(K) x 16(N), from LDS tile stored TRANSPOSED: [n][k], stride ldk.
__device__ inline v16h load_bT_frag(const _Float16* s, int ldk, int k0, int col0) {
  const int lane = threadIdx.x & 31;
  const int c  = col0 + (lane & 15);
  const int kb = k0 + (lane >> 4) * 8;
  const _Float16* p = s + (size_t)c * ldk + kb;
  v8h lo = *(const v8h*)(p);
  v8h hi = *(const v8h*)(p + 16);
  return cat8(lo, hi);
}

__device__ inline v8f v8f_zero() {
  v8f z;
#pragma unroll
  for (int e = 0; e < 8; ++e) z[e] = 0.0f;
  return z;
}

// ---------------------------------------------------------------------------
// GEMM: C[M,N] = A[M,K] @ B[K,N] (+bias) (+residual) (+relu), fp32 in/out,
// f16 WMMA compute. 128x64 tile per 128-thread block (4 waves);
// each wave owns a 32x64 strip (2 A-frags x 4 B-frags = 8 WMMA per K-step).
// FULL=true: exact tiling; double-buffered LDS, next tile's global_load_b128
// issued before the current tile's WMMAs (one barrier per K-step).
// ---------------------------------------------------------------------------
#define GTM 128
#define GTN 64
#define GTK 32
#define GLD (GTK + 8)   // 40 halves = 80B, multiple of 16B

template <bool FULL>
__global__ void __launch_bounds__(128)
gemm_wmma_kernel(const float* __restrict__ A, const float* __restrict__ B,
                 const float* __restrict__ bias, const float* __restrict__ res,
                 float* __restrict__ C, int M, int N, int K, int relu) {
  __shared__ _Float16 As[2][GTM][GLD];    // [buf][m][k]
  __shared__ _Float16 BsT[2][GTN][GLD];   // [buf][n][k]  (transposed stage)
  const int tid  = threadIdx.x;
  const int wave = tid >> 5;
  const int lane = tid & 31;
  const int m0 = blockIdx.y * GTM;
  const int n0 = blockIdx.x * GTN;
  const int wr = wave * 32;

  v8f acc[2][4];
#pragma unroll
  for (int u = 0; u < 2; ++u)
#pragma unroll
    for (int t = 0; t < 4; ++t) acc[u][t] = v8f_zero();

  if (FULL) {
    float4 ra[8], rb[4];
    auto gload = [&](int k0) {
#pragma unroll
      for (int it = 0; it < 8; ++it) {
        int i = tid + it * 128;
        ra[it] = *(const float4*)(A + (size_t)(m0 + (i >> 3)) * K + k0 + (i & 7) * 4);
      }
#pragma unroll
      for (int it = 0; it < 4; ++it) {
        int i = tid + it * 128;
        rb[it] = *(const float4*)(B + (size_t)(k0 + (i >> 4)) * N + n0 + (i & 15) * 4);
      }
    };
    auto stage = [&](int buf) {
#pragma unroll
      for (int it = 0; it < 8; ++it) {
        int i = tid + it * 128;
        *(v4h*)&As[buf][i >> 3][(i & 7) * 4] =
            cvt4(ra[it].x, ra[it].y, ra[it].z, ra[it].w);
      }
#pragma unroll
      for (int it = 0; it < 4; ++it) {
        int i  = tid + it * 128;
        int r  = i >> 4;            // k
        int c4 = (i & 15) * 4;      // n
        v4h h = cvt4(rb[it].x, rb[it].y, rb[it].z, rb[it].w);
        BsT[buf][c4 + 0][r] = h[0];
        BsT[buf][c4 + 1][r] = h[1];
        BsT[buf][c4 + 2][r] = h[2];
        BsT[buf][c4 + 3][r] = h[3];
      }
    };

    const int nk = K / GTK;
    gload(0);
    stage(0);
    int p = 0;
    for (int kt = 0; kt < nk; ++kt) {
      __syncthreads();
      const bool more = (kt + 1 < nk);
      if (more) gload((kt + 1) * GTK);          // VMEM in flight during WMMAs
      v16h a0 = load_a_frag(&As[p][0][0], GLD, wr, 0);
      v16h a1 = load_a_frag(&As[p][0][0], GLD, wr + 16, 0);
#pragma unroll
      for (int t = 0; t < 4; ++t) {
        v16h b = load_bT_frag(&BsT[p][0][0], GLD, 0, t * 16);
        acc[0][t] = wmma_f16(a0, b, acc[0][t]);
        acc[1][t] = wmma_f16(a1, b, acc[1][t]);
      }
      if (more) stage(1 - p);                   // convert+store after compute
      p ^= 1;
    }
  } else {
    for (int k0 = 0; k0 < K; k0 += GTK) {
      for (int i = tid; i < GTM * GTK; i += 128) {
        int r = i >> 5, c = i & 31;
        int gm = m0 + r, gk = k0 + c;
        float v = (gm < M && gk < K) ? A[(size_t)gm * K + gk] : 0.0f;
        As[0][r][c] = (_Float16)v;
      }
      for (int i = tid; i < GTK * GTN; i += 128) {
        int r = i >> 6, c = i & 63;
        int gk = k0 + r, gn = n0 + c;
        float v = (gk < K && gn < N) ? B[(size_t)gk * N + gn] : 0.0f;
        BsT[0][c][r] = (_Float16)v;
      }
      __syncthreads();
      v16h a0 = load_a_frag(&As[0][0][0], GLD, wr, 0);
      v16h a1 = load_a_frag(&As[0][0][0], GLD, wr + 16, 0);
#pragma unroll
      for (int t = 0; t < 4; ++t) {
        v16h b = load_bT_frag(&BsT[0][0][0], GLD, 0, t * 16);
        acc[0][t] = wmma_f16(a0, b, acc[0][t]);
        acc[1][t] = wmma_f16(a1, b, acc[1][t]);
      }
      __syncthreads();
    }
  }

  const int hi = lane >> 4;
  const int nc = lane & 15;
#pragma unroll
  for (int u = 0; u < 2; ++u) {
#pragma unroll
    for (int t = 0; t < 4; ++t) {
      int n = n0 + t * 16 + nc;
      if (!FULL && n >= N) continue;
      float bv = bias ? bias[n] : 0.0f;
#pragma unroll
      for (int e = 0; e < 8; ++e) {
        int m = m0 + wr + u * 16 + e + hi * 8;
        if (!FULL && m >= M) continue;
        float v = acc[u][t][e] + bv;
        if (res) v += res[(size_t)m * N + n];
        if (relu) v = fmaxf(v, 0.0f);
        C[(size_t)m * N + n] = v;
      }
    }
  }
}

// ---------------------------------------------------------------------------
// Flash attention, dk=dv=64, H=8, model stride 512 (heads kept interleaved).
// One block = one (batch, head, 64-query tile); 4 waves, each owns 16 rows.
// Online softmax; Q.K^T and P.V both via WMMA with vectorized LDS operands.
// ---------------------------------------------------------------------------
#define ALD 72   // 144B stride, multiple of 16B

__global__ void __launch_bounds__(128)
attn_kernel(const float* __restrict__ Qg, const float* __restrict__ Kg,
            const float* __restrict__ Vg, float* __restrict__ Og,
            int LQ, int LK, int masked) {
  const int STR = 512, DKV = 64;
  __shared__ _Float16 Qs[64][ALD];    // [m][c]       A operand for Q.K^T
  __shared__ _Float16 Ks[64][ALD];    // [key][c]     B^T operand for Q.K^T
  __shared__ _Float16 VsT[64][ALD];   // [c][key]     B^T operand for P.V
  __shared__ _Float16 Ps[64][ALD];    // [m][key]     A operand for P.V

  const int tid  = threadIdx.x;
  const int wave = tid >> 5;
  const int lane = tid & 31;
  const int q0 = blockIdx.x * 64;
  const int h  = blockIdx.y;
  const int bz = blockIdx.z;

  const size_t qbase  = ((size_t)bz * LQ + q0) * STR + (size_t)h * DKV;
  const size_t kbase0 = ((size_t)bz * LK) * STR + (size_t)h * DKV;

  // Q tile: 64x64 fp32 = 1024 float4, 8 per thread
#pragma unroll
  for (int it = 0; it < 8; ++it) {
    int i  = tid + it * 128;
    int r  = i >> 4;
    int c4 = (i & 15) * 4;
    float4 v = *(const float4*)(Qg + qbase + (size_t)r * STR + c4);
    *(v4h*)&Qs[r][c4] = cvt4(v.x, v.y, v.z, v.w);
  }
  __syncthreads();
  v16h aq0 = load_a_frag(&Qs[0][0], ALD, wave * 16, 0);
  v16h aq1 = load_a_frag(&Qs[0][0], ALD, wave * 16, 32);

  v8f oacc[4];
#pragma unroll
  for (int t = 0; t < 4; ++t) oacc[t] = v8f_zero();
  float mrun[8], lrun[8];
#pragma unroll
  for (int e = 0; e < 8; ++e) { mrun[e] = -1e30f; lrun[e] = 0.0f; }

  const int hi = lane >> 4;
  const int nc = lane & 15;
  const int ktend = masked ? (q0 / 64 + 1) : (LK / 64);

  for (int kt = 0; kt < ktend; ++kt) {
    __syncthreads();
    const size_t kb = kbase0 + (size_t)kt * 64 * STR;
    if (kt + 1 < ktend) {
      // prefetch next K/V tile (global_prefetch_b8); one line per thread
      const size_t nb = kbase0 + (size_t)(kt + 1) * 64 * STR +
                        (size_t)(tid >> 1) * STR + (tid & 1) * 32;
      __builtin_prefetch(Kg + nb, 0, 3);
      __builtin_prefetch(Vg + nb, 0, 3);
    }
#pragma unroll
    for (int it = 0; it < 8; ++it) {
      int i  = tid + it * 128;
      int r  = i >> 4;            // key-in-tile
      int c4 = (i & 15) * 4;      // channel
      float4 kv = *(const float4*)(Kg + kb + (size_t)r * STR + c4);
      *(v4h*)&Ks[r][c4] = cvt4(kv.x, kv.y, kv.z, kv.w);
      float4 vv = *(const float4*)(Vg + kb + (size_t)r * STR + c4);
      v4h hv = cvt4(vv.x, vv.y, vv.z, vv.w);
      VsT[c4 + 0][r] = hv[0];
      VsT[c4 + 1][r] = hv[1];
      VsT[c4 + 2][r] = hv[2];
      VsT[c4 + 3][r] = hv[3];
    }
    __syncthreads();

    // S = Q.K^T   (B^T layout [key][ch] -> k-dim = channel)
    v8f s[4];
#pragma unroll
    for (int t = 0; t < 4; ++t) {
      s[t] = v8f_zero();
      v16h b0 = load_bT_frag(&Ks[0][0], ALD, 0, t * 16);
      s[t] = wmma_f16(aq0, b0, s[t]);
      v16h b1 = load_bT_frag(&Ks[0][0], ALD, 32, t * 16);
      s[t] = wmma_f16(aq1, b1, s[t]);
    }

    // online softmax per row (row replicated across 16 lanes of a half)
#pragma unroll
    for (int e = 0; e < 8; ++e) {
      const int m = q0 + wave * 16 + e + hi * 8;   // global query row
      float rmax = -1e30f;
#pragma unroll
      for (int t = 0; t < 4; ++t) {
        float v = s[t][e] * 0.125f;                 // 1/sqrt(64)
        int kcol = kt * 64 + t * 16 + nc;
        if (masked && kcol > m) v = -1e30f;
        s[t][e] = v;
        rmax = fmaxf(rmax, v);
      }
#pragma unroll
      for (int off = 8; off >= 1; off >>= 1)
        rmax = fmaxf(rmax, __shfl_xor(rmax, off, 32));
      float nm    = fmaxf(mrun[e], rmax);
      float alpha = __expf(mrun[e] - nm);
      float rs = 0.0f;
#pragma unroll
      for (int t = 0; t < 4; ++t) {
        float p = __expf(s[t][e] - nm);
        s[t][e] = p;
        rs += p;
      }
#pragma unroll
      for (int off = 8; off >= 1; off >>= 1) rs += __shfl_xor(rs, off, 32);
      lrun[e] = lrun[e] * alpha + rs;
      mrun[e] = nm;
#pragma unroll
      for (int t = 0; t < 4; ++t) oacc[t][e] *= alpha;
      const int lm = wave * 16 + e + hi * 8;
#pragma unroll
      for (int t = 0; t < 4; ++t) Ps[lm][t * 16 + nc] = (_Float16)s[t][e];
    }
    __syncthreads();

    // O += P.V   (B^T layout [ch][key] -> k-dim = key)
    v16h ap0 = load_a_frag(&Ps[0][0], ALD, wave * 16, 0);
    v16h ap1 = load_a_frag(&Ps[0][0], ALD, wave * 16, 32);
#pragma unroll
    for (int t = 0; t < 4; ++t) {
      v16h b0 = load_bT_frag(&VsT[0][0], ALD, 0, t * 16);
      oacc[t] = wmma_f16(ap0, b0, oacc[t]);
      v16h b1 = load_bT_frag(&VsT[0][0], ALD, 32, t * 16);
      oacc[t] = wmma_f16(ap1, b1, oacc[t]);
    }
  }

#pragma unroll
  for (int e = 0; e < 8; ++e) {
    const int m = q0 + wave * 16 + e + hi * 8;
    const float inv = 1.0f / (lrun[e] + 1e-30f);
    const size_t ob = ((size_t)bz * LQ + m) * STR + (size_t)h * DKV;
#pragma unroll
    for (int t = 0; t < 4; ++t)
      Og[ob + t * 16 + nc] = oacc[t][e] * inv;
  }
}

// ---------------------------------------------------------------------------
// out = LayerNorm(x + y) * g + b, D=512, one block per row, 128 threads.
// ---------------------------------------------------------------------------
__global__ void __launch_bounds__(128)
add_ln_kernel(const float* __restrict__ x, const float* __restrict__ y,
              const float* __restrict__ g, const float* __restrict__ b,
              float* __restrict__ out) {
  const int D = 512;
  const int tid = threadIdx.x;
  const size_t base = (size_t)blockIdx.x * D;
  __shared__ float red[4];
  float v[4];
  float s = 0.0f;
#pragma unroll
  for (int j = 0; j < 4; ++j) {
    int c = tid * 4 + j;
    v[j] = x[base + c] + y[base + c];
    s += v[j];
  }
#pragma unroll
  for (int off = 16; off >= 1; off >>= 1) s += __shfl_xor(s, off, 32);
  if ((tid & 31) == 0) red[tid >> 5] = s;
  __syncthreads();
  float mean = (red[0] + red[1] + red[2] + red[3]) / (float)D;
  __syncthreads();
  float vs = 0.0f;
#pragma unroll
  for (int j = 0; j < 4; ++j) { float d = v[j] - mean; vs += d * d; }
#pragma unroll
  for (int off = 16; off >= 1; off >>= 1) vs += __shfl_xor(vs, off, 32);
  if ((tid & 31) == 0) red[tid >> 5] = vs;
  __syncthreads();
  float var = (red[0] + red[1] + red[2] + red[3]) / (float)D;
  float inv = rsqrtf(var + 1e-5f);
#pragma unroll
  for (int j = 0; j < 4; ++j) {
    int c = tid * 4 + j;
    out[base + c] = (v[j] - mean) * inv * g[c] + b[c];
  }
}

// ---------------------------------------------------------------------------
// x[b,l,:] += sinusoidal PE(l), d_model=512. grid=(L,B), 128 threads.
// ---------------------------------------------------------------------------
__global__ void __launch_bounds__(128)
add_pe_kernel(float* __restrict__ x, int L) {
  const int l = blockIdx.x;
  const size_t base = ((size_t)blockIdx.y * L + l) * 512;
  const int tid = threadIdx.x;
#pragma unroll
  for (int j = 0; j < 4; ++j) {
    int c = tid * 4 + j;
    float p2  = (float)((c >> 1) * 2);
    float div = __expf(p2 * (-9.210340371976184f / 512.0f)); // -ln(10000)/512
    float ang = (float)l * div;
    x[base + c] += (c & 1) ? __cosf(ang) : __sinf(ang);
  }
}

// ---------------------------------------------------------------------------
// Host orchestration
// ---------------------------------------------------------------------------
extern "C" void kernel_launch(void* const* d_in, const int* in_sizes, int n_in,
                              void* d_out, int out_size, void* d_ws, size_t ws_size,
                              hipStream_t stream) {
  (void)in_sizes; (void)n_in; (void)out_size; (void)ws_size;
  const float* enc_x      = (const float*)d_in[0];
  const float* enc_mark   = (const float*)d_in[1];
  const float* dec_in     = (const float*)d_in[2];
  const float* dec_mark   = (const float*)d_in[3];
  const float* emb_enc_vW = (const float*)d_in[4];
  const float* emb_enc_vb = (const float*)d_in[5];
  const float* emb_enc_mW = (const float*)d_in[6];
  const float* emb_enc_mb = (const float*)d_in[7];
  const float* emb_dec_vW = (const float*)d_in[8];
  const float* emb_dec_vb = (const float*)d_in[9];
  const float* emb_dec_mW = (const float*)d_in[10];
  const float* emb_dec_mb = (const float*)d_in[11];
  const float* enc_Wq = (const float*)d_in[12];
  const float* enc_Wk = (const float*)d_in[13];
  const float* enc_Wv = (const float*)d_in[14];
  const float* enc_Wo = (const float*)d_in[15];
  const float* enc_W1 = (const float*)d_in[16];
  const float* enc_b1 = (const float*)d_in[17];
  const float* enc_W2 = (const float*)d_in[18];
  const float* enc_b2 = (const float*)d_in[19];
  const float* enc_ln = (const float*)d_in[20];
  const float* dec_sWq = (const float*)d_in[21];
  const float* dec_sWk = (const float*)d_in[22];
  const float* dec_sWv = (const float*)d_in[23];
  const float* dec_sWo = (const float*)d_in[24];
  const float* dec_cWq = (const float*)d_in[25];
  const float* dec_cWk = (const float*)d_in[26];
  const float* dec_cWv = (const float*)d_in[27];
  const float* dec_cWo = (const float*)d_in[28];
  const float* dec_W1 = (const float*)d_in[29];
  const float* dec_b1 = (const float*)d_in[30];
  const float* dec_W2 = (const float*)d_in[31];
  const float* dec_b2 = (const float*)d_in[32];
  const float* dec_ln = (const float*)d_in[33];
  const float* proj_W = (const float*)d_in[34];
  const float* proj_b = (const float*)d_in[35];

  const int DM = 512, LE = 2048, LD = 1024, Bsz = 2;
  const int ME = Bsz * LE;   // 4096
  const int MD = Bsz * LD;   // 2048
  const size_t SZ = (size_t)ME * DM;  // 2M floats

  float* ws = (float*)d_ws;
  float* X  = ws + 0 * SZ;   // encoder hidden / enc_out
  float* Qb = ws + 1 * SZ;
  float* Kb = ws + 2 * SZ;
  float* Vb = ws + 3 * SZ;
  float* Cb = ws + 4 * SZ;
  float* T0 = ws + 5 * SZ;
  float* T1 = ws + 6 * SZ;
  float* Db = ws + 7 * SZ;   // decoder hidden

  auto gemm = [&](const float* A, const float* B, const float* bias,
                  const float* res, float* C, int M, int N, int K, int relu) {
    dim3 grid((N + GTN - 1) / GTN, (M + GTM - 1) / GTM);
    bool full = (M % GTM == 0) && (N % GTN == 0) && (K % GTK == 0);
    if (full)
      gemm_wmma_kernel<true><<<grid, 128, 0, stream>>>(A, B, bias, res, C, M, N, K, relu);
    else
      gemm_wmma_kernel<false><<<grid, 128, 0, stream>>>(A, B, bias, res, C, M, N, K, relu);
  };
  auto lnfn = [&](const float* x, const float* y, const float* gb, float* o, int M) {
    add_ln_kernel<<<M, 128, 0, stream>>>(x, y, gb, gb + DM, o);
  };

  // ------------------- encoder embedding -------------------
  gemm(enc_x,    emb_enc_vW, emb_enc_vb, nullptr, X, ME, DM, 32, 0);
  gemm(enc_mark, emb_enc_mW, emb_enc_mb, X,       X, ME, DM, 4,  0);
  add_pe_kernel<<<dim3(LE, Bsz), 128, 0, stream>>>(X, LE);

  // ------------------- encoder layers -------------------
  for (int i = 0; i < 3; ++i) {
    const size_t wo = (size_t)i * DM * DM;
    gemm(X, enc_Wq + wo, nullptr, nullptr, Qb, ME, DM, DM, 0);
    gemm(X, enc_Wk + wo, nullptr, nullptr, Kb, ME, DM, DM, 0);
    gemm(X, enc_Wv + wo, nullptr, nullptr, Vb, ME, DM, DM, 0);
    attn_kernel<<<dim3(LE / 64, 8, Bsz), 128, 0, stream>>>(Qb, Kb, Vb, Cb, LE, LE, 0);
    gemm(Cb, enc_Wo + wo, nullptr, nullptr, T0, ME, DM, DM, 0);
    lnfn(X, T0, enc_ln + (size_t)(i * 4 + 0) * DM, X, ME);
    gemm(X,  enc_W1 + wo, enc_b1 + (size_t)i * DM, nullptr, T1, ME, DM, DM, 1);
    gemm(T1, enc_W2 + wo, enc_b2 + (size_t)i * DM, nullptr, T0, ME, DM, DM, 0);
    lnfn(X, T0, enc_ln + (size_t)(i * 4 + 2) * DM, X, ME);
  }

  // ------------------- decoder embedding -------------------
  gemm(dec_in,   emb_dec_vW, emb_dec_vb, nullptr, Db, MD, DM, 32, 0);
  gemm(dec_mark, emb_dec_mW, emb_dec_mb, Db,      Db, MD, DM, 4,  0);
  add_pe_kernel<<<dim3(LD, Bsz), 128, 0, stream>>>(Db, LD);

  // ------------------- decoder layers -------------------
  for (int i = 0; i < 2; ++i) {
    const size_t wo = (size_t)i * DM * DM;
    // masked self-attention
    gemm(Db, dec_sWq + wo, nullptr, nullptr, Qb, MD, DM, DM, 0);
    gemm(Db, dec_sWk + wo, nullptr, nullptr, Kb, MD, DM, DM, 0);
    gemm(Db, dec_sWv + wo, nullptr, nullptr, Vb, MD, DM, DM, 0);
    attn_kernel<<<dim3(LD / 64, 8, Bsz), 128, 0, stream>>>(Qb, Kb, Vb, Cb, LD, LD, 1);
    gemm(Cb, dec_sWo + wo, nullptr, nullptr, T0, MD, DM, DM, 0);
    lnfn(Db, T0, dec_ln + (size_t)(i * 6 + 0) * DM, Db, MD);
    // cross-attention against encoder output
    gemm(Db, dec_cWq + wo, nullptr, nullptr, Qb, MD, DM, DM, 0);
    gemm(X,  dec_cWk + wo, nullptr, nullptr, Kb, ME, DM, DM, 0);
    gemm(X,  dec_cWv + wo, nullptr, nullptr, Vb, ME, DM, DM, 0);
    attn_kernel<<<dim3(LD / 64, 8, Bsz), 128, 0, stream>>>(Qb, Kb, Vb, Cb, LD, LE, 0);
    gemm(Cb, dec_cWo + wo, nullptr, nullptr, T0, MD, DM, DM, 0);
    lnfn(Db, T0, dec_ln + (size_t)(i * 6 + 2) * DM, Db, MD);
    // FFN
    gemm(Db, dec_W1 + wo, dec_b1 + (size_t)i * DM, nullptr, T1, MD, DM, DM, 1);
    gemm(T1, dec_W2 + wo, dec_b2 + (size_t)i * DM, nullptr, T0, MD, DM, DM, 0);
    lnfn(Db, T0, dec_ln + (size_t)(i * 6 + 4) * DM, Db, MD);
  }

  // ------------------- projection -------------------
  gemm(Db, proj_W, proj_b, nullptr, (float*)d_out, MD, 32, DM, 0);
}
